// ORGAN_39041252721106
// MI455X (gfx1250) — matrix operand order repaired and checked by reading, not compiled, not run on hardware
//
#include <hip/hip_runtime.h>
#include <hip/hip_bf16.h>
#include <stdint.h>

// ---------------- CDNA5 WMMA types ----------------
typedef __attribute__((ext_vector_type(16))) __bf16 v16bf;
typedef __attribute__((ext_vector_type(8)))  __bf16 v8bf;
typedef __attribute__((ext_vector_type(8)))  float  v8f;

#define BATCH   2048
#define EMB     64
#define HID     1024
#define VOCAB   32
#define MAXLEN  100
#define SEQW    (MAXLEN + 1)
#define GATES_N (4 * HID)   // 4096
#define TOK_PAD 0
#define TOK_BOS 1
#define TOK_EOS 2

// fp32 -> bf16 (round to nearest even)
__device__ __forceinline__ unsigned short f2bf(float x) {
    unsigned int u = __float_as_uint(x);
    unsigned int r = u + 0x7FFFu + ((u >> 16) & 1u);
    return (unsigned short)(r >> 16);
}

__device__ __forceinline__ float sigf(float x) { return 1.0f / (1.0f + __expf(-x)); }

// ---------------------------------------------------------------------------
// Weight prep: W [Nrows x K] row-major fp32 -> bf16 swizzled WMMA-B layout.
// (k, n): kt=k/32, kin=k%32, lane=(kin/16)*16 | (n%16), e=kin%16
// dst = (kt*(Nrows/16) + n/16)*512 + lane*16 + e
// => one contiguous 32B v16bf per lane per 32-k slab.
// ---------------------------------------------------------------------------
__global__ void k_prep_w(const float* __restrict__ W, unsigned short* __restrict__ Bsw,
                         int Nrows, int K) {
    int idx = blockIdx.x * blockDim.x + threadIdx.x;
    if (idx >= Nrows * K) return;
    int nrow = idx / K;
    int k    = idx - nrow * K;
    int kt = k >> 5, kin = k & 31;
    int lane = ((kin >> 4) << 4) | (nrow & 15);
    int e = kin & 15;
    size_t dst = (((size_t)kt * (size_t)(Nrows >> 4) + (size_t)(nrow >> 4)) << 9)
               + (size_t)(lane << 4) + (size_t)e;
    Bsw[dst] = f2bf(W[idx]);
}

__global__ void k_bias_add(const float* __restrict__ a, const float* __restrict__ b,
                           float* __restrict__ out, int n) {
    int i = blockIdx.x * blockDim.x + threadIdx.x;
    if (i < n) out[i] = a[i] + b[i];
}

// ---------------------------------------------------------------------------
// Init: zero cell/hidden state, seed prev tokens, write seq column 0.
// ---------------------------------------------------------------------------
__global__ void k_init(float* c0, float* c1, float* h1f,
                       unsigned short* h0bf, unsigned short* h1bf,
                       const int* __restrict__ prevs_in,
                       int* prev, int* is_end, int* lengths,
                       float* out_seq) {
    int i = blockIdx.x * blockDim.x + threadIdx.x;
    if (i < BATCH * HID) {
        c0[i] = 0.0f; c1[i] = 0.0f; h1f[i] = 0.0f;
        h0bf[i] = 0;  h1bf[i] = 0;
    }
    if (i < BATCH) {
        int p = prevs_in[i];
        prev[i]    = p;
        is_end[i]  = (p == TOK_EOS) ? 1 : 0;
        lengths[i] = 0;
        out_seq[(size_t)i * SEQW] = (float)p;
    }
}

// ---------------------------------------------------------------------------
// Embedding gather -> bf16 activations [BATCH x EMB]
// ---------------------------------------------------------------------------
__global__ void k_embed(const float* __restrict__ emb, const int* __restrict__ prev,
                        unsigned short* __restrict__ xbf) {
    int i = blockIdx.x * blockDim.x + threadIdx.x;
    if (i >= BATCH * EMB) return;
    int n = i >> 6;
    int e = i & (EMB - 1);
    xbf[i] = f2bf(emb[prev[n] * EMB + e]);
}

// ---------------------------------------------------------------------------
// Fused dual-pass WMMA GEMM + LSTM cell:
//   gates = A1@B1 + A2@B2 + bias ; then i,f,g,o -> c,h in-register.
// Block: 256 threads (8 wave32); tile = 32 batch rows x 128 hidden cols.
// Each wave owns one 16-wide hidden tile and accumulates ALL FOUR gate tiles
// (cols jt, jt+1024, jt+2048, jt+3072) for 2 M-subtiles -> 8 WMMAs / 32-k slab.
// A slabs double-buffered in LDS via CDNA5 GLOBAL_LOAD_ASYNC_TO_LDS_B128.
// ---------------------------------------------------------------------------
__global__ __launch_bounds__(256) void k_gemm_cell(
    const unsigned short* __restrict__ A1, const unsigned short* __restrict__ B1, int K1,
    const unsigned short* __restrict__ A2, const unsigned short* __restrict__ B2, int K2,
    const float* __restrict__ bias,
    float* __restrict__ c, float* __restrict__ hf32, unsigned short* __restrict__ hbf)
{
    __shared__ __align__(16) unsigned short As[2][32 * 32];  // double-buffered A slab

    const int tid    = threadIdx.x;
    const int wave   = tid >> 5;
    const int lane   = tid & 31;
    const int mbase  = blockIdx.y * 32;
    const int jtile  = blockIdx.x * 8 + wave;        // hidden col tile, 0..63
    const int mlocal = lane & 15;
    const int khalf  = lane >> 4;
    const int nColTiles = GATES_N >> 4;              // 256
    // LDS byte offset of As (generic pointer: low 32 bits = group offset)
    const unsigned asBase = (unsigned)(size_t)(&As[0][0]);

    v8f acc[2][4];
    #pragma unroll
    for (int mt = 0; mt < 2; ++mt)
        #pragma unroll
        for (int g = 0; g < 4; ++g)
            #pragma unroll
            for (int r = 0; r < 8; ++r) acc[mt][g][r] = 0.0f;

    // async stage of one 32x32 bf16 A slab into LDS buffer `buf`
    auto stage = [&](const unsigned short* A, int K, int slab, int buf) {
        if (tid < 128) {                             // wave-uniform predicate
            int row = tid >> 2;                      // 0..31
            int kc  = (tid & 3) << 3;                // 0,8,16,24
            const unsigned short* src =
                A + (size_t)(mbase + row) * (size_t)K + (size_t)((slab << 5) + kc);
            unsigned ldsOff = asBase + (unsigned)(buf * 2048 + (((row << 5) + kc) << 1));
            unsigned long long ga = (unsigned long long)(size_t)src;
            asm volatile("global_load_async_to_lds_b128 %0, %1, off"
                         :: "v"(ldsOff), "v"(ga) : "memory");
        }
    };

    #pragma unroll 1
    for (int pass = 0; pass < 2; ++pass) {
        const unsigned short* A = pass ? A2 : A1;
        const unsigned short* B = pass ? B2 : B1;
        const int K = pass ? K2 : K1;
        const int nslab = K >> 5;

        __syncthreads();                             // LDS buffers free (prev pass done)
        stage(A, K, 0, 0);

        for (int i = 0; i < nslab; ++i) {
            const int cur = i & 1;
            if (i) __syncthreads();                  // readers of buf[cur^1] done
            if (i + 1 < nslab) {
                stage(A, K, i + 1, cur ^ 1);         // prefetch next slab (async)
                asm volatile("s_wait_asynccnt 0x1" ::: "memory");  // slab i retired
            } else {
                asm volatile("s_wait_asynccnt 0x0" ::: "memory");
            }
            __syncthreads();                         // slab i visible to all waves

            // B fragments: 4 gates, one contiguous v16bf per lane each
            v16bf bfrag[4];
            #pragma unroll
            for (int g = 0; g < 4; ++g) {
                size_t ntile = (size_t)g * (HID >> 4) + (size_t)jtile;  // g*64 + jt
                const unsigned short* bp =
                    B + (((size_t)i * nColTiles + ntile) << 9) + (size_t)(lane << 4);
                bfrag[g] = *reinterpret_cast<const v16bf*>(bp);
                if (i + 1 < nslab)
                    __builtin_prefetch(bp + (size_t)nColTiles * 512, 0, 3);
            }

            #pragma unroll
            for (int mt = 0; mt < 2; ++mt) {
                // A fragment: contiguous k order (k = 16*khalf + s), same as B prep
                const unsigned short* ap =
                &As[cur][(((mt << 4) + mlocal) << 5) + (khalf << 4)];
            v8bf lo = *reinterpret_cast<const v8bf*>(ap);
            v8bf hi = *reinterpret_cast<const v8bf*>(ap + 8);
                v16bf afrag;
                #pragma unroll
                for (int q = 0; q < 8; ++q) { afrag[q] = lo[q]; afrag[8 + q] = hi[q]; }
                #pragma unroll
                for (int g = 0; g < 4; ++g)
                    acc[mt][g] = __builtin_amdgcn_wmma_f32_16x16x32_bf16(
                        false, afrag, false, bfrag[g], (short)0, acc[mt][g], false, false);
            }
        }
    }

    // ---- fused LSTM cell epilogue (C layout: m = r + 8*khalf, n = lane&15) ----
    const int jcol = (jtile << 4) + mlocal;
    const float bi = bias[jcol];
    const float bf_ = bias[HID + jcol];
    const float bg = bias[2 * HID + jcol];
    const float bo = bias[3 * HID + jcol];
    #pragma unroll
    for (int mt = 0; mt < 2; ++mt) {
        #pragma unroll
        for (int r = 0; r < 8; ++r) {
            int row = mbase + (mt << 4) + (khalf << 3) + r;
            size_t ix = (size_t)row * HID + (size_t)jcol;
            float ig = acc[mt][0][r] + bi;
            float fg = acc[mt][1][r] + bf_;
            float gg = acc[mt][2][r] + bg;
            float og = acc[mt][3][r] + bo;
            float cn = sigf(fg) * c[ix] + sigf(ig) * tanhf(gg);
            float hn = sigf(og) * tanhf(cn);
            c[ix]   = cn;
            hf32[ix] = hn;
            hbf[ix] = f2bf(hn);
        }
    }
}

// ---------------------------------------------------------------------------
// Logits (2048x32 @ 1024) + Gumbel-max sampling + EOS bookkeeping.
// ---------------------------------------------------------------------------
__global__ __launch_bounds__(256) void k_sample(
    const float* __restrict__ h1, const float* __restrict__ Wout,
    const float* __restrict__ bout, int* __restrict__ prev,
    int* __restrict__ is_end, int* __restrict__ lengths,
    float* __restrict__ out_seq, int t)
{
    const int wave = threadIdx.x >> 5;
    const int lane = threadIdx.x & 31;
    const int n = blockIdx.x * 8 + wave;
    if (n >= BATCH) return;

    float acc = bout[lane];
    const float* hr = h1 + (size_t)n * HID;
    const float* wr = Wout + (size_t)lane * HID;
    #pragma unroll 4
    for (int k = 0; k < HID; k += 4) {
        float4 hv = *reinterpret_cast<const float4*>(hr + k);
        float4 wv = *reinterpret_cast<const float4*>(wr + k);
        acc += hv.x * wv.x + hv.y * wv.y + hv.z * wv.z + hv.w * wv.w;
    }

    unsigned int s = 0x9E3779B9u * (unsigned)(t + 1)
                   ^ 0x85EBCA6Bu * (unsigned)(n + 1)
                   ^ 0xC2B2AE35u * (unsigned)(lane + 17);
    s ^= s >> 16; s *= 0x7FEB352Du; s ^= s >> 15; s *= 0x846CA68Bu; s ^= s >> 16;
    float u = fmaxf((float)(s >> 8) * (1.0f / 16777216.0f), 1e-12f);
    float val = acc - __logf(fmaxf(-__logf(u), 1e-20f));

    int best = lane; float bv = val;
    #pragma unroll
    for (int off = 16; off > 0; off >>= 1) {
        float ov = __shfl_xor(bv, off, 32);
        int   ob = __shfl_xor(best, off, 32);
        if (ov > bv || (ov == bv && ob < best)) { bv = ov; best = ob; }
    }

    if (lane == 0) {
        int end = is_end[n];
        int cur = end ? TOK_PAD : best;
        lengths[n] += end ? 0 : 1;
        is_end[n] = end | (cur == TOK_EOS);
        prev[n] = cur;
        out_seq[(size_t)n * SEQW + (size_t)(t + 1)] = (float)cur;
    }
}

__global__ void k_lengths(const int* __restrict__ lengths, float* __restrict__ out_len) {
    int i = blockIdx.x * blockDim.x + threadIdx.x;
    if (i < BATCH) out_len[i] = (float)(lengths[i] + 1);
}

// ---------------------------------------------------------------------------
// Host launcher
// ---------------------------------------------------------------------------
extern "C" void kernel_launch(void* const* d_in, const int* in_sizes, int n_in,
                              void* d_out, int out_size, void* d_ws, size_t ws_size,
                              hipStream_t stream) {
    (void)in_sizes; (void)n_in; (void)out_size; (void)ws_size;

    const int*   prevs = (const int*)  d_in[0];
    const float* emb   = (const float*)d_in[1];
    const float* Wih0  = (const float*)d_in[2];
    const float* Whh0  = (const float*)d_in[3];
    const float* bih0  = (const float*)d_in[4];
    const float* bhh0  = (const float*)d_in[5];
    const float* Wih1  = (const float*)d_in[6];
    const float* Whh1  = (const float*)d_in[7];
    const float* bih1  = (const float*)d_in[8];
    const float* bhh1  = (const float*)d_in[9];
    const float* Wout  = (const float*)d_in[10];
    const float* bout  = (const float*)d_in[11];

    float* out_seq = (float*)d_out;
    float* out_len = out_seq + (size_t)BATCH * SEQW;

    size_t off = 0;
    auto carve = [&](size_t bytes) -> char* {
        char* p = (char*)d_ws + off;
        off += (bytes + 255) & ~(size_t)255;
        return p;
    };
    unsigned short* wih0sw = (unsigned short*)carve((size_t)GATES_N * EMB * 2);
    unsigned short* whh0sw = (unsigned short*)carve((size_t)GATES_N * HID * 2);
    unsigned short* wih1sw = (unsigned short*)carve((size_t)GATES_N * HID * 2);
    unsigned short* whh1sw = (unsigned short*)carve((size_t)GATES_N * HID * 2);
    float* bias0 = (float*)carve((size_t)GATES_N * 4);
    float* bias1 = (float*)carve((size_t)GATES_N * 4);
    float* c0   = (float*)carve((size_t)BATCH * HID * 4);
    float* c1   = (float*)carve((size_t)BATCH * HID * 4);
    float* h0f  = (float*)carve((size_t)BATCH * HID * 4);   // scratch (layer0 h fp32)
    float* h1f  = (float*)carve((size_t)BATCH * HID * 4);   // read by sampler
    unsigned short* h0bf[2];
    unsigned short* h1bf[2];
    h0bf[0] = (unsigned short*)carve((size_t)BATCH * HID * 2);
    h0bf[1] = (unsigned short*)carve((size_t)BATCH * HID * 2);
    h1bf[0] = (unsigned short*)carve((size_t)BATCH * HID * 2);
    h1bf[1] = (unsigned short*)carve((size_t)BATCH * HID * 2);
    unsigned short* xbf = (unsigned short*)carve((size_t)BATCH * EMB * 2);
    int* prev    = (int*)carve((size_t)BATCH * 4);
    int* is_end  = (int*)carve((size_t)BATCH * 4);
    int* lengths = (int*)carve((size_t)BATCH * 4);

    const int TPB = 256;

    // ---- weight prep ----
    k_prep_w<<<(GATES_N * EMB + TPB - 1) / TPB, TPB, 0, stream>>>(Wih0, wih0sw, GATES_N, EMB);
    k_prep_w<<<(GATES_N * HID + TPB - 1) / TPB, TPB, 0, stream>>>(Whh0, whh0sw, GATES_N, HID);
    k_prep_w<<<(GATES_N * HID + TPB - 1) / TPB, TPB, 0, stream>>>(Wih1, wih1sw, GATES_N, HID);
    k_prep_w<<<(GATES_N * HID + TPB - 1) / TPB, TPB, 0, stream>>>(Whh1, whh1sw, GATES_N, HID);
    k_bias_add<<<(GATES_N + TPB - 1) / TPB, TPB, 0, stream>>>(bih0, bhh0, bias0, GATES_N);
    k_bias_add<<<(GATES_N + TPB - 1) / TPB, TPB, 0, stream>>>(bih1, bhh1, bias1, GATES_N);
    k_init<<<(BATCH * HID + TPB - 1) / TPB, TPB, 0, stream>>>(
        c0, c1, h1f, h0bf[0], h1bf[0], prevs, prev, is_end, lengths, out_seq);

    // ---- 100 sequential decode steps, h buffers ping-ponged across steps ----
    dim3 gemmGrid(HID / 128, BATCH / 32);   // (8, 64)
    for (int t = 0; t < MAXLEN; ++t) {
        const int pr = t & 1, pw = pr ^ 1;

        k_embed<<<(BATCH * EMB + TPB - 1) / TPB, TPB, 0, stream>>>(emb, prev, xbf);

        k_gemm_cell<<<gemmGrid, TPB, 0, stream>>>(
            xbf, wih0sw, EMB, h0bf[pr], whh0sw, HID, bias0, c0, h0f, h0bf[pw]);

        k_gemm_cell<<<gemmGrid, TPB, 0, stream>>>(
            h0bf[pw], wih1sw, HID, h1bf[pr], whh1sw, HID, bias1, c1, h1f, h1bf[pw]);

        k_sample<<<BATCH / 8, TPB, 0, stream>>>(
            h1f, Wout, bout, prev, is_end, lengths, out_seq, t);
    }

    k_lengths<<<(BATCH + TPB - 1) / TPB, TPB, 0, stream>>>(lengths, out_len);
}
